// LocationSensitiveAttention_30210799960234
// MI455X (gfx1250) — matrix operand order repaired and checked by reading, not compiled, not run on hardware
//
#include <hip/hip_runtime.h>
#include <hip/hip_bf16.h>

#define B_     64
#define T_     2048
#define ENC_   512
#define DEC_   1024
#define ATT_   128
#define NFILT_ 32
#define KSIZE_ 31
#define PAD_   15

typedef __attribute__((ext_vector_type(16))) __bf16         bf16x16;
typedef __attribute__((ext_vector_type(8)))  float          f32x8;

struct Bytes32 { uint4 lo, hi; };

// Native fp32 -> bf16 (RNE). Plain fptrunc lets the backend use the
// packed hardware converter (v_cvt_pk_bf16_f32) instead of ALU bit-twiddling.
__device__ __forceinline__ __bf16 f2bf(float f) { return (__bf16)f; }

// Native V_TANH_F32 when the builtin exists; library tanhf otherwise.
__device__ __forceinline__ float fast_tanh(float x) {
#if __has_builtin(__builtin_amdgcn_tanhf)
  return __builtin_amdgcn_tanhf(x);
#elif __has_builtin(__builtin_amdgcn_tanh_f32)
  return __builtin_amdgcn_tanh_f32(x);
#else
  return tanhf(x);
#endif
}

// ---------------------------------------------------------------------------
// P1: M[k,a] = sum_f conv_w[f,k] * loc_dense[f,a], stored transposed [a][32] bf16
// (k = 31 zero-padded so the location term is a single K=32 WMMA step)
// ---------------------------------------------------------------------------
__global__ void k_prep_mt(const float* __restrict__ conv_w,
                          const float* __restrict__ loc_dense,
                          __bf16* __restrict__ mt) {
  int idx = blockIdx.x * blockDim.x + threadIdx.x;       // 0..4095
  if (idx >= ATT_ * 32) return;
  int a = idx >> 5, k = idx & 31;
  float s = 0.f;
  if (k < KSIZE_) {
    for (int f = 0; f < NFILT_; ++f)
      s += conv_w[f * KSIZE_ + k] * loc_dense[f * ATT_ + a];
  }
  mt[a * 32 + k] = f2bf(s);
}

// ---------------------------------------------------------------------------
// P2: W1 [ENC,ATT] fp32  ->  W1t [ATT][ENC] bf16 (column-major for B fragments)
// ---------------------------------------------------------------------------
__global__ void k_prep_w1t(const float* __restrict__ W1,
                           __bf16* __restrict__ w1t) {
  int idx = blockIdx.x * blockDim.x + threadIdx.x;       // 0..65535
  if (idx >= ENC_ * ATT_) return;
  int e = idx >> 7, a = idx & 127;                       // coalesced read of W1
  w1t[a * ENC_ + e] = f2bf(W1[idx]);
}

// ---------------------------------------------------------------------------
// P3: proj_query[b,a] = sum_d query[b,d] * W2[d,a]
// ---------------------------------------------------------------------------
__global__ void k_proj_query(const float* __restrict__ q,
                             const float* __restrict__ W2,
                             float* __restrict__ pq) {
  int b = blockIdx.x, a = threadIdx.x;                   // block = 128 threads
  const float* qb = q + b * DEC_;
  float s = 0.f;
#pragma unroll 4
  for (int d = 0; d < DEC_; ++d)
    s += qb[d] * W2[d * ATT_ + a];
  pq[b * ATT_ + a] = s;
}

// ---------------------------------------------------------------------------
// Main: per (b, 128-t tile): scores[b,t] = V . tanh(keys@W1 + pw(conv)@M + pq)
// 8 waves/block; each wave owns 16 t-rows x 128 a-cols = 8 WMMA tiles.
// bf16 WMMA, fp32 accumulate. 17 k-steps (1 location + 16 over ENC).
// ---------------------------------------------------------------------------
__global__ void __launch_bounds__(256) k_scores(
    const float* __restrict__ keys, const float* __restrict__ pw,
    const int* __restrict__ mask, const float* __restrict__ Vv,
    const __bf16* __restrict__ w1t, const __bf16* __restrict__ mt,
    const float* __restrict__ pq, float* __restrict__ scores) {
  const int b = blockIdx.y;
  const int tid = threadIdx.x;
  const int wave = tid >> 5;
  const int lane = tid & 31;
  const int hi = lane >> 4;          // k-group selector (ISA A/B layouts)
  const int lanelo = lane & 15;      // A: row-in-tile, B: col-in-tile
  const int tbase = blockIdx.x * 128 + wave * 16;
  const int trow = tbase + lanelo;

  f32x8 acc[8];
#pragma unroll
  for (int nt = 0; nt < 8; ++nt)
#pragma unroll
    for (int v = 0; v < 8; ++v) acc[nt][v] = 0.f;

  // ---- location term: A = im2col window of prev_weights (16x32), B = Mt ----
  {
    bf16x16 af;
#pragma unroll
    for (int i = 0; i < 16; ++i) {
      // A 16-bit layout: k = [8*hi .. +7] (regs0-3), [16+8*hi .. +7] (regs4-7)
      int k = (i < 8 ? i : i + 8) + 8 * hi;
      int tt = trow + k - PAD_;
      float v = (k < KSIZE_ && tt >= 0 && tt < T_) ? pw[b * T_ + tt] : 0.f;
      af[i] = f2bf(v);
    }
#pragma unroll
    for (int nt = 0; nt < 8; ++nt) {
      // B 16-bit layout: lane = col, K = [8*hi .. +7], [16+8*hi .. +7]
      const uint4* qp = (const uint4*)(mt + (nt * 16 + lanelo) * 32);
      Bytes32 raw{qp[hi], qp[2 + hi]};
      bf16x16 bf = __builtin_bit_cast(bf16x16, raw);
      acc[nt] = __builtin_amdgcn_wmma_f32_16x16x32_bf16(
          false, af, false, bf, (short)0, acc[nt], false, false);
    }
  }

  // ---- key projection: keys[b, t, :] @ W1, 16 steps of K=32 ----
  const float* arow = keys + ((size_t)b * T_ + trow) * ENC_;
  for (int k0 = 0; k0 < ENC_; k0 += 32) {
    const float* p0 = arow + k0 + 8 * hi;
    float4 c0 = *(const float4*)(p0);
    float4 c1 = *(const float4*)(p0 + 4);
    float4 c2 = *(const float4*)(p0 + 16);
    float4 c3 = *(const float4*)(p0 + 20);
    bf16x16 af;
    af[0] = f2bf(c0.x);  af[1] = f2bf(c0.y);  af[2] = f2bf(c0.z);  af[3] = f2bf(c0.w);
    af[4] = f2bf(c1.x);  af[5] = f2bf(c1.y);  af[6] = f2bf(c1.z);  af[7] = f2bf(c1.w);
    af[8] = f2bf(c2.x);  af[9] = f2bf(c2.y);  af[10] = f2bf(c2.z); af[11] = f2bf(c2.w);
    af[12] = f2bf(c3.x); af[13] = f2bf(c3.y); af[14] = f2bf(c3.z); af[15] = f2bf(c3.w);
#pragma unroll
    for (int nt = 0; nt < 8; ++nt) {
      const uint4* qp = (const uint4*)(w1t + (size_t)(nt * 16 + lanelo) * ENC_ + k0);
      Bytes32 raw{qp[hi], qp[2 + hi]};
      bf16x16 bf = __builtin_bit_cast(bf16x16, raw);
      acc[nt] = __builtin_amdgcn_wmma_f32_16x16x32_bf16(
          false, af, false, bf, (short)0, acc[nt], false, false);
    }
  }

  // ---- epilogue: + proj_query, tanh, dot with V, reduce over a ----
  // D layout: element (v, lane) = (row v + 8*hi, col lane&15)
  float partial[8];
#pragma unroll
  for (int v = 0; v < 8; ++v) partial[v] = 0.f;
#pragma unroll
  for (int nt = 0; nt < 8; ++nt) {
    int a = nt * 16 + lanelo;
    float pqa = pq[b * ATT_ + a];
    float va = Vv[a];
#pragma unroll
    for (int v = 0; v < 8; ++v)
      partial[v] += fast_tanh(acc[nt][v] + pqa) * va;
  }
  // reduce across the 16 lanes of each half (xor < 16 stays within a half)
#pragma unroll
  for (int off = 1; off < 16; off <<= 1)
#pragma unroll
    for (int v = 0; v < 8; ++v)
      partial[v] += __shfl_xor(partial[v], off, 32);

  if (lanelo == 0) {
#pragma unroll
    for (int v = 0; v < 8; ++v) {
      int t = tbase + v + 8 * hi;
      float s = partial[v];
      if (mask[b * T_ + t] == 0) s = -1e9f;
      scores[b * T_ + t] = s;
    }
  }
}

// ---------------------------------------------------------------------------
// Softmax over T per batch row
// ---------------------------------------------------------------------------
__global__ void k_softmax(const float* __restrict__ scores,
                          float* __restrict__ weights) {
  __shared__ float red[256];
  int b = blockIdx.x, tid = threadIdx.x;
  const float* s = scores + b * T_;
  float m = -3.4e38f;
  for (int t = tid; t < T_; t += 256) m = fmaxf(m, s[t]);
  red[tid] = m;
  __syncthreads();
  for (int o = 128; o > 0; o >>= 1) {
    if (tid < o) red[tid] = fmaxf(red[tid], red[tid + o]);
    __syncthreads();
  }
  float mx = red[0];
  __syncthreads();
  float sum = 0.f;
  for (int t = tid; t < T_; t += 256) sum += __expf(s[t] - mx);
  red[tid] = sum;
  __syncthreads();
  for (int o = 128; o > 0; o >>= 1) {
    if (tid < o) red[tid] += red[tid + o];
    __syncthreads();
  }
  float inv = 1.f / red[0];
  for (int t = tid; t < T_; t += 256) weights[b * T_ + t] = __expf(s[t] - mx) * inv;
}

// ---------------------------------------------------------------------------
// Context: partial sums over 256-row T-chunks (deterministic, no atomics)
// ---------------------------------------------------------------------------
__global__ void k_ctx_partial(const float* __restrict__ keys,
                              const float* __restrict__ w,
                              float* __restrict__ part) {
  __shared__ float wl[256];
  int b = blockIdx.y, chunk = blockIdx.x, tid = threadIdx.x;  // block = 128
  int t0 = chunk * 256;
  wl[tid] = w[b * T_ + t0 + tid];
  wl[tid + 128] = w[b * T_ + t0 + tid + 128];
  __syncthreads();
  const float* kb = keys + ((size_t)b * T_ + t0) * ENC_ + tid * 4;
  float4 acc = {0.f, 0.f, 0.f, 0.f};
#pragma unroll 4
  for (int j = 0; j < 256; ++j) {
    float4 kv = *(const float4*)(kb + (size_t)j * ENC_);
    float ww = wl[j];
    acc.x += ww * kv.x; acc.y += ww * kv.y;
    acc.z += ww * kv.z; acc.w += ww * kv.w;
  }
  float* o = part + (size_t)(chunk * B_ + b) * ENC_ + tid * 4;
  *(float4*)o = acc;
}

__global__ void k_ctx_reduce(const float* __restrict__ part,
                             float* __restrict__ out) {
  int idx = blockIdx.x * 256 + threadIdx.x;  // B*ENC = 32768
  float s = 0.f;
#pragma unroll
  for (int c = 0; c < 8; ++c) s += part[(size_t)c * (B_ * ENC_) + idx];
  out[idx] = s;
}

// ---------------------------------------------------------------------------
extern "C" void kernel_launch(void* const* d_in, const int* in_sizes, int n_in,
                              void* d_out, int out_size, void* d_ws, size_t ws_size,
                              hipStream_t stream) {
  const float* query     = (const float*)d_in[0];  // [64,1,1024]
  const float* keys      = (const float*)d_in[1];  // [64,2048,512]
  const float* pw        = (const float*)d_in[2];  // [64,2048]
  const int*   mask      = (const int*)d_in[3];    // [64,2048]
  const float* W1        = (const float*)d_in[4];  // [512,128]
  const float* W2        = (const float*)d_in[5];  // [1024,128]
  const float* Vv        = (const float*)d_in[6];  // [128]
  const float* conv_w    = (const float*)d_in[7];  // [32,1,31]
  const float* loc_dense = (const float*)d_in[8];  // [32,128]

  float* out_ctx = (float*)d_out;            // [B,1,ENC] = 32768 floats
  float* out_w   = out_ctx + B_ * ENC_;      // [B,T]     = 131072 floats

  // workspace layout (~1.7 MB)
  float* ws_scores = (float*)d_ws;                           // B*T
  float* ws_pq     = ws_scores + B_ * T_;                    // B*ATT
  float* ws_part   = ws_pq + B_ * ATT_;                      // 8*B*ENC
  __bf16* ws_w1t   = (__bf16*)(ws_part + 8 * B_ * ENC_);     // ATT*ENC bf16
  __bf16* ws_mt    = ws_w1t + ATT_ * ENC_;                   // ATT*32 bf16

  k_prep_mt<<<16, 256, 0, stream>>>(conv_w, loc_dense, ws_mt);
  k_prep_w1t<<<256, 256, 0, stream>>>(W1, ws_w1t);
  k_proj_query<<<B_, 128, 0, stream>>>(query, W2, ws_pq);
  k_scores<<<dim3(T_ / 128, B_), 256, 0, stream>>>(keys, pw, mask, Vv,
                                                   ws_w1t, ws_mt, ws_pq, ws_scores);
  k_softmax<<<B_, 256, 0, stream>>>(ws_scores, out_w);
  k_ctx_partial<<<dim3(8, B_), 128, 0, stream>>>(keys, out_w, ws_part);
  k_ctx_reduce<<<128, 256, 0, stream>>>(ws_part, out_ctx);
}
